// MusicDiffusionTransformer_8332236554622
// MI455X (gfx1250) — matrix operand order repaired and checked by reading, not compiled, not run on hardware
//
#include <hip/hip_runtime.h>
#include <hip/hip_bf16.h>

// ---------------------------------------------------------------------------
// MusicDiffusionTransformer on gfx1250 (MI455X)
// bf16 WMMA GEMMs (pipelined b128 fragment loads) + flash attention
// ---------------------------------------------------------------------------

typedef __attribute__((ext_vector_type(16))) __bf16          v16bf;
typedef __attribute__((ext_vector_type(8)))  float           v8f;
typedef __attribute__((ext_vector_type(8)))  unsigned short  us8;

union bfpack {            // two 16-byte chunks <-> one WMMA bf16 fragment
    us8   u8[2];
    v16bf v;
};

#define NTOK 1088
#define DIMD 512
#define MLPD 2048

static __device__ __forceinline__ unsigned short f32_to_bf16_raw(float f) {
    unsigned u = __builtin_bit_cast(unsigned, f);
    unsigned short h = (unsigned short)((u + 0x7FFFu + ((u >> 16) & 1u)) >> 16);
    return h;                                   // round-to-nearest-even
}
static __device__ __forceinline__ float bf2f(unsigned short r) {
    unsigned u = ((unsigned)r) << 16;
    return __builtin_bit_cast(float, u);
}
static __device__ __forceinline__ float gelu_exact(float x) {
    return 0.5f * x * (1.0f + erff(x * 0.70710678118654752f));
}

// ---------------------------------------------------------------------------
// WMMA GEMM: out = act(Abf16[M,K] @ Wt_bf16[N,K]^T + bias [+resid] [+rowvec])
// one wave per 16x64 output block (4 N-tiles share each A fragment)
// A fragment (bf16 16x32, same K order as B): lane holds row=lane&15; K chunks
//   [h*16 .. h*16+7] and [h*16+8 .. h*16+15]  -> two contiguous b128 loads
// B fragment: lane holds col=lane&15; K = h*16+i -> 16 contiguous bf16 in W^T
// K-loop is software-pipelined: fragments for step k+1 are issued before the
// WMMAs of step k so loads stay outstanding across WMMA issue.
// ---------------------------------------------------------------------------
__global__ void gemm_wmma_kernel(const unsigned short* __restrict__ Abf,
                                 const unsigned short* __restrict__ Wt,
                                 const float* __restrict__ bias,
                                 const float* __restrict__ resid,
                                 const float* __restrict__ rowvec, int rowvec_div,
                                 float* __restrict__ outF,
                                 unsigned short* __restrict__ outBF,
                                 int M, int N, int K, int act)
{
    const int lane = threadIdx.x & 31;
    const int hlf  = lane >> 4;
    const int c16  = lane & 15;
    const int row0 = blockIdx.y * 16;
    const int col0 = blockIdx.x * 64;

    v8f acc[4] = {{}, {}, {}, {}};
    const unsigned short* arow = Abf + (size_t)(row0 + c16) * K;
    const unsigned short* bcol[4];
#pragma unroll
    for (int nt = 0; nt < 4; ++nt)
        bcol[nt] = Wt + (size_t)(col0 + nt * 16 + c16) * K;

    // prologue: fragments for k0 = 0
    bfpack a_cur, b_cur[4];
    a_cur.u8[0] = *(const us8*)(arow + hlf * 16);
    a_cur.u8[1] = *(const us8*)(arow + hlf * 16 + 8);
#pragma unroll
    for (int nt = 0; nt < 4; ++nt) {
        const unsigned short* bp = bcol[nt] + hlf * 16;
        b_cur[nt].u8[0] = *(const us8*)(bp);
        b_cur[nt].u8[1] = *(const us8*)(bp + 8);
    }

    // steady state: issue next step's loads, then this step's WMMAs
    int k0 = 0;
    for (; k0 + 32 < K; k0 += 32) {
        const int kn = k0 + 32;
        if (kn + 32 < K) {
            __builtin_prefetch(arow + kn + 32, 0, 1);
            __builtin_prefetch(bcol[0] + kn + 32, 0, 1);
        }
        bfpack a_nxt, b_nxt[4];
        a_nxt.u8[0] = *(const us8*)(arow + kn + hlf * 16);
        a_nxt.u8[1] = *(const us8*)(arow + kn + hlf * 16 + 8);
#pragma unroll
        for (int nt = 0; nt < 4; ++nt) {
            const unsigned short* bp = bcol[nt] + kn + hlf * 16;
            b_nxt[nt].u8[0] = *(const us8*)(bp);
            b_nxt[nt].u8[1] = *(const us8*)(bp + 8);
        }
#pragma unroll
        for (int nt = 0; nt < 4; ++nt)
            acc[nt] = __builtin_amdgcn_wmma_f32_16x16x32_bf16(
                false, a_cur.v, false, b_cur[nt].v, (short)0, acc[nt], false, false);
        a_cur = a_nxt;
#pragma unroll
        for (int nt = 0; nt < 4; ++nt) b_cur[nt] = b_nxt[nt];
    }
    // drain
#pragma unroll
    for (int nt = 0; nt < 4; ++nt)
        acc[nt] = __builtin_amdgcn_wmma_f32_16x16x32_bf16(
            false, a_cur.v, false, b_cur[nt].v, (short)0, acc[nt], false, false);

#pragma unroll
    for (int nt = 0; nt < 4; ++nt) {
#pragma unroll
        for (int i = 0; i < 8; ++i) {              // C layout: row=i+8h, col=c16
            int r = row0 + i + 8 * hlf;
            int c = col0 + nt * 16 + c16;
            float v = acc[nt][i];
            if (bias)   v += bias[c];
            if (resid)  v += resid[(size_t)r * N + c];
            if (rowvec) v += rowvec[(size_t)(r / rowvec_div) * N + c];
            if (act == 1) v = gelu_exact(v);
            size_t idx = (size_t)r * N + c;
            if (outF)  outF[idx]  = v;
            if (outBF) outBF[idx] = f32_to_bf16_raw(v);
        }
    }
}

// ---------------------------------------------------------------------------
// transpose+convert weights: Wt_bf16[n*K+k] = bf16(W_f32[k*N+n])
// ---------------------------------------------------------------------------
__global__ void wconv_kernel(const float* __restrict__ W,
                             unsigned short* __restrict__ Wt, int K, int N)
{
    int id = blockIdx.x * blockDim.x + threadIdx.x;
    if (id >= K * N) return;
    int k = id % K, n = id / K;
    Wt[id] = f32_to_bf16_raw(W[(size_t)k * N + n]);
}

// ---------------------------------------------------------------------------
// Flash-style multi-query attention: one wave per (batch, head, 16-query tile)
// q_bf : [2,1088,512] bf16 (heads concat)
// kv_bf: [2,1088,128] bf16 (K | V)
// vt_bf: [2,64,1088]  bf16 (V transposed for contiguous B-fragments)
// obf  : [2,1088,512] bf16 heads-concat output
// ---------------------------------------------------------------------------
__global__ void attention_kernel(const unsigned short* __restrict__ q,
                                 const unsigned short* __restrict__ kvb,
                                 const unsigned short* __restrict__ vt,
                                 unsigned short* __restrict__ obf)
{
    const int lane = threadIdx.x & 31;
    const int hlf  = lane >> 4;
    const int c16  = lane & 15;
    const int qt   = blockIdx.x;     // 0..67
    const int hd   = blockIdx.y;     // 0..7
    const int b    = blockIdx.z;     // 0..1
    const float scale = 0.125f;      // 64^-0.5

    __shared__ unsigned short p_lds[16][32];

    // Q tile, A-layout (16 rows x 64 dims -> two 16x32 bf16 fragments)
    const unsigned short* qrow =
        q + ((size_t)b * NTOK + qt * 16 + c16) * DIMD + hd * 64;
    bfpack aq0, aq1;
    aq0.u8[0] = *(const us8*)(qrow + hlf * 16);
    aq0.u8[1] = *(const us8*)(qrow + 8 + hlf * 16);
    aq1.u8[0] = *(const us8*)(qrow + 32 + hlf * 16);
    aq1.u8[1] = *(const us8*)(qrow + 40 + hlf * 16);

    v8f o0 = {}, o1 = {}, o2 = {}, o3 = {};
    float m[8], l[8];
#pragma unroll
    for (int i = 0; i < 8; ++i) { m[i] = -3.0e38f; l[i] = 0.f; }

    for (int kp = 0; kp < NTOK / 32; ++kp) {           // 34 key pairs
        int key0 = kp * 32;
        v8f s0 = {}, s1 = {};
        // S = Q @ K^T  (B fragment: col=key, K-dim=head dim, contiguous bf16)
        {
            const unsigned short* kr = kvb + ((size_t)b * NTOK + key0 + c16) * 128 + hlf * 16;
            bfpack bk0, bk1;
            bk0.u8[0] = *(const us8*)(kr);
            bk0.u8[1] = *(const us8*)(kr + 8);
            bk1.u8[0] = *(const us8*)(kr + 32);
            bk1.u8[1] = *(const us8*)(kr + 40);
            s0 = __builtin_amdgcn_wmma_f32_16x16x32_bf16(false, aq0.v, false, bk0.v, (short)0, s0, false, false);
            s0 = __builtin_amdgcn_wmma_f32_16x16x32_bf16(false, aq1.v, false, bk1.v, (short)0, s0, false, false);
        }
        {
            const unsigned short* kr = kvb + ((size_t)b * NTOK + key0 + 16 + c16) * 128 + hlf * 16;
            bfpack bk0, bk1;
            bk0.u8[0] = *(const us8*)(kr);
            bk0.u8[1] = *(const us8*)(kr + 8);
            bk1.u8[0] = *(const us8*)(kr + 32);
            bk1.u8[1] = *(const us8*)(kr + 40);
            s1 = __builtin_amdgcn_wmma_f32_16x16x32_bf16(false, aq0.v, false, bk0.v, (short)0, s1, false, false);
            s1 = __builtin_amdgcn_wmma_f32_16x16x32_bf16(false, aq1.v, false, bk1.v, (short)0, s1, false, false);
        }

        // online softmax (row r=i+8h lives in lane-half h; butterfly over 16 lanes)
#pragma unroll
        for (int i = 0; i < 8; ++i) {
            float v0 = s0[i] * scale, v1 = s1[i] * scale;
            float rm = fmaxf(v0, v1);
            rm = fmaxf(rm, __shfl_xor(rm, 1));
            rm = fmaxf(rm, __shfl_xor(rm, 2));
            rm = fmaxf(rm, __shfl_xor(rm, 4));
            rm = fmaxf(rm, __shfl_xor(rm, 8));
            float mn   = fmaxf(m[i], rm);
            float corr = __expf(m[i] - mn);
            float e0 = __expf(v0 - mn), e1 = __expf(v1 - mn);
            float rs = e0 + e1;
            rs += __shfl_xor(rs, 1);
            rs += __shfl_xor(rs, 2);
            rs += __shfl_xor(rs, 4);
            rs += __shfl_xor(rs, 8);
            l[i] = l[i] * corr + rs;
            m[i] = mn;
            o0[i] *= corr; o1[i] *= corr; o2[i] *= corr; o3[i] *= corr;
            int prow = i + 8 * hlf;
            p_lds[prow][c16]      = f32_to_bf16_raw(e0);
            p_lds[prow][16 + c16] = f32_to_bf16_raw(e1);
        }
        __syncthreads();

        // reload P in A-layout (two contiguous b128 chunks per lane)
        bfpack ap;
        ap.u8[0] = *(const us8*)(&p_lds[c16][hlf * 16]);
        ap.u8[1] = *(const us8*)(&p_lds[c16][hlf * 16 + 8]);

        // O += P @ V   (V^T fragments contiguous in key)
#pragma unroll
        for (int nt = 0; nt < 4; ++nt) {
            const unsigned short* vp =
                vt + ((size_t)b * 64 + nt * 16 + c16) * NTOK + key0 + hlf * 16;
            bfpack bv;
            bv.u8[0] = *(const us8*)(vp);
            bv.u8[1] = *(const us8*)(vp + 8);
            v8f* op = (nt == 0) ? &o0 : (nt == 1) ? &o1 : (nt == 2) ? &o2 : &o3;
            *op = __builtin_amdgcn_wmma_f32_16x16x32_bf16(
                false, ap.v, false, bv.v, (short)0, *op, false, false);
        }
        __syncthreads();
    }

    // normalize + store heads-concat bf16
#pragma unroll
    for (int nt = 0; nt < 4; ++nt) {
        const v8f oo = (nt == 0) ? o0 : (nt == 1) ? o1 : (nt == 2) ? o2 : o3;
#pragma unroll
        for (int i = 0; i < 8; ++i) {
            int row = qt * 16 + i + 8 * hlf;
            float v = oo[i] / l[i];
            obf[((size_t)b * NTOK + row) * DIMD + hd * 64 + nt * 16 + c16] =
                f32_to_bf16_raw(v);
        }
    }
}

// V transpose: vt[b][d][key] = kv_bf[b][key][64+d]
__global__ void vtrans_kernel(const unsigned short* __restrict__ kvb,
                              unsigned short* __restrict__ vt)
{
    int id = blockIdx.x * blockDim.x + threadIdx.x;
    if (id >= 2 * 64 * NTOK) return;
    int key = id % NTOK, d = (id / NTOK) % 64, b = id / (NTOK * 64);
    vt[id] = kvb[((size_t)b * NTOK + key) * 128 + 64 + d];
}

// ---------------------------------------------------------------------------
// LayerNorm: one 128-thread block per token row; emits f32 + bf16 mirror
// ---------------------------------------------------------------------------
__global__ void layernorm_kernel(const float* __restrict__ in,
                                 const float* __restrict__ g,
                                 const float* __restrict__ bb,
                                 float* __restrict__ outF,
                                 unsigned short* __restrict__ outBF,
                                 int D)
{
    int row = blockIdx.x;
    const float* x = in + (size_t)row * D;
    __shared__ float red[128];

    float s = 0.f;
    for (int i = threadIdx.x; i < D; i += blockDim.x) s += x[i];
    red[threadIdx.x] = s; __syncthreads();
    for (int off = 64; off > 0; off >>= 1) {
        if ((int)threadIdx.x < off) red[threadIdx.x] += red[threadIdx.x + off];
        __syncthreads();
    }
    float mean = red[0] / (float)D;
    __syncthreads();

    float v = 0.f;
    for (int i = threadIdx.x; i < D; i += blockDim.x) {
        float d = x[i] - mean; v += d * d;
    }
    red[threadIdx.x] = v; __syncthreads();
    for (int off = 64; off > 0; off >>= 1) {
        if ((int)threadIdx.x < off) red[threadIdx.x] += red[threadIdx.x + off];
        __syncthreads();
    }
    float rstd = rsqrtf(red[0] / (float)D + 1e-5f);

    for (int i = threadIdx.x; i < D; i += blockDim.x) {
        float y = (x[i] - mean) * rstd * g[i] + bb[i];
        outF[(size_t)row * D + i] = y;
        if (outBF) outBF[(size_t)row * D + i] = f32_to_bf16_raw(y);
    }
}

// ---------------------------------------------------------------------------
// Scalar paths: convs, M=2 time-MLPs, N=1 output head
// ---------------------------------------------------------------------------
__global__ void gemm_scalar_kernel(const float* __restrict__ A,
                                   const float* __restrict__ W,
                                   const float* __restrict__ bias,
                                   float* __restrict__ out,
                                   int M, int N, int K, int act)
{
    int id = blockIdx.x * blockDim.x + threadIdx.x;
    if (id >= M * N) return;
    int mm = id / N, nn = id % N;
    float s = bias ? bias[nn] : 0.f;
    for (int k = 0; k < K; ++k)
        s += A[(size_t)mm * K + k] * W[(size_t)k * N + nn];
    if (act == 1) s = gelu_exact(s);
    out[id] = s;
}

__global__ void conv1_kernel(const float* __restrict__ x,
                             const float* __restrict__ w,
                             const float* __restrict__ bias,
                             float* __restrict__ out)
{
    int id = blockIdx.x * blockDim.x + threadIdx.x;
    if (id >= 2 * 256 * 1024) return;
    int t = id % 1024, oc = (id / 1024) % 256, b = id / (1024 * 256);
    float s = bias[oc];
#pragma unroll
    for (int j = 0; j < 7; ++j) {
        int ti = t + j - 3;
        if (ti >= 0 && ti < 1024) s += x[b * 1024 + ti] * w[oc * 7 + j];
    }
    out[id] = fmaxf(s, 0.f);
}

__global__ void conv2_kernel(const float* __restrict__ in,
                             const float* __restrict__ w,
                             const float* __restrict__ bias,
                             float* __restrict__ out)
{
    int id = blockIdx.x * blockDim.x + threadIdx.x;
    if (id >= 2 * 512 * 1024) return;
    int t = id % 1024, oc = (id / 1024) % 512, b = id / (1024 * 512);
    float s = bias[oc];
    for (int ic = 0; ic < 256; ++ic) {
        const float* xr = in + ((size_t)b * 256 + ic) * 1024;
        const float* wr = w + ((size_t)oc * 256 + ic) * 7;
#pragma unroll
        for (int j = 0; j < 7; ++j) {
            int ti = t + j - 3;
            if (ti >= 0 && ti < 1024) s += xr[ti] * wr[j];
        }
    }
    out[id] = fmaxf(s, 0.f);
}

__global__ void pack_kernel(const float* __restrict__ conv_out, // [2,512,1024]
                            const float* __restrict__ cond,     // [2,64,512]
                            float* __restrict__ h)              // [2,1088,512]
{
    int id = blockIdx.x * blockDim.x + threadIdx.x;
    if (id >= 2 * NTOK * DIMD) return;
    int c = id % DIMD, t = (id / DIMD) % NTOK, b = id / (DIMD * NTOK);
    float v = (t < 1024)
        ? conv_out[((size_t)b * DIMD + c) * 1024 + t]
        : cond[((size_t)b * 64 + (t - 1024)) * DIMD + c];
    h[id] = v;
}

__global__ void timefeat_kernel(const float* __restrict__ ts, float* __restrict__ te)
{
    int id = blockIdx.x * blockDim.x + threadIdx.x;
    if (id >= 2 * DIMD) return;
    int j = id % DIMD, b = id / DIMD;
    int half = DIMD / 2;
    int jj = (j < half) ? j : j - half;
    float freq = __expf((float)jj * (-9.210340371976184f / 255.0f));
    float a = ts[b] * freq;
    te[id] = (j < half) ? __sinf(a) : __cosf(a);
}

__global__ void final_out_kernel(const unsigned short* __restrict__ mid, // [2,1088,1024] bf16
                                 const float* __restrict__ w2,           // [1024,1]
                                 const float* __restrict__ b2,
                                 float* __restrict__ out)                // [2,1024]
{
    int id = blockIdx.x * blockDim.x + threadIdx.x;
    if (id >= 2 * 1024) return;
    int t = id % 1024, b = id / 1024;
    const unsigned short* m = mid + ((size_t)b * NTOK + t) * 1024;
    float s = b2[0];
    for (int k = 0; k < 1024; ++k) s += bf2f(m[k]) * w2[k];
    out[id] = s;
}

// ---------------------------------------------------------------------------
// Host side
// ---------------------------------------------------------------------------
static void launch_wconv(const float* W, unsigned short* Wt, int K, int N,
                         hipStream_t s)
{
    wconv_kernel<<<(K * N + 255) / 256, 256, 0, s>>>(W, Wt, K, N);
}

static void launch_gemm(const unsigned short* A, const unsigned short* Wt,
                        const float* bias, const float* resid,
                        const float* rowvec, int rowdiv,
                        float* outF, unsigned short* outBF,
                        int M, int N, int K, int act, hipStream_t s)
{
    dim3 grid(N / 64, M / 16);
    gemm_wmma_kernel<<<grid, 32, 0, s>>>(A, Wt, bias, resid, rowvec,
                                         rowdiv ? rowdiv : 1, outF, outBF,
                                         M, N, K, act);
}

extern "C" void kernel_launch(void* const* d_in, const int* in_sizes, int n_in,
                              void* d_out, int out_size, void* d_ws, size_t ws_size,
                              hipStream_t stream)
{
    (void)in_sizes; (void)n_in; (void)out_size; (void)ws_size;

    const float* x        = (const float*)d_in[0];
    const float* tsteps   = (const float*)d_in[1];
    const float* cond     = (const float*)d_in[2];
    const float* conv_w1  = (const float*)d_in[3];
    const float* conv_b1  = (const float*)d_in[4];
    const float* conv_w2  = (const float*)d_in[5];
    const float* conv_b2  = (const float*)d_in[6];
    const float* tpe_w1   = (const float*)d_in[7];
    const float* tpe_b1   = (const float*)d_in[8];
    const float* tpe_w2   = (const float*)d_in[9];
    const float* tpe_b2   = (const float*)d_in[10];
    const float* ln1_g    = (const float*)d_in[11];
    const float* ln1_b    = (const float*)d_in[12];
    const float* wq       = (const float*)d_in[13];
    const float* wkv      = (const float*)d_in[14];
    const float* wo       = (const float*)d_in[15];
    const float* bo       = (const float*)d_in[16];
    const float* ln2_g    = (const float*)d_in[17];
    const float* ln2_b    = (const float*)d_in[18];
    const float* ffn_w1   = (const float*)d_in[19];
    const float* ffn_b1   = (const float*)d_in[20];
    const float* ffn_w2   = (const float*)d_in[21];
    const float* ffn_b2   = (const float*)d_in[22];
    const float* tm_w1    = (const float*)d_in[23];
    const float* tm_b1    = (const float*)d_in[24];
    const float* tm_w2    = (const float*)d_in[25];
    const float* tm_b2    = (const float*)d_in[26];
    const float* fn_g     = (const float*)d_in[27];
    const float* fn_b     = (const float*)d_in[28];
    const float* out_w1   = (const float*)d_in[29];
    const float* out_b1   = (const float*)d_in[30];
    const float* out_w2   = (const float*)d_in[31];
    const float* out_b2   = (const float*)d_in[32];

    char* ws = (char*)d_ws;
    size_t off = 0;
    auto alloc = [&](size_t bytes) -> void* {
        void* p = ws + off;
        off = (off + bytes + 255) & ~(size_t)255;
        return p;
    };
    const int M = 2 * NTOK;                               // 2176 rows
    float*          conv1o  = (float*)alloc((size_t)2 * 256 * 1024 * 4);
    float*          conv2o  = (float*)alloc((size_t)2 * 512 * 1024 * 4);
    float*          hbuf    = (float*)alloc((size_t)M * DIMD * 4);
    float*          hx      = (float*)alloc((size_t)M * DIMD * 4);
    unsigned short* hx_bf   = (unsigned short*)alloc((size_t)M * DIMD * 2);
    unsigned short* q_bf    = (unsigned short*)alloc((size_t)M * DIMD * 2);
    unsigned short* kv_bf   = (unsigned short*)alloc((size_t)M * 128 * 2);
    unsigned short* vt_bf   = (unsigned short*)alloc((size_t)2 * 64 * NTOK * 2);
    unsigned short* o_bf    = (unsigned short*)alloc((size_t)M * DIMD * 2);
    unsigned short* ffn_bf  = (unsigned short*)alloc((size_t)M * MLPD * 2);
    unsigned short* omid_bf = (unsigned short*)alloc((size_t)M * 1024 * 2);
    unsigned short* wt_buf  = (unsigned short*)alloc((size_t)DIMD * MLPD * 2); // largest weight
    float*          te      = (float*)alloc((size_t)2 * DIMD * 4);
    float*          tpe_mid = (float*)alloc((size_t)2 * MLPD * 4);
    float*          temb    = (float*)alloc((size_t)2 * DIMD * 4);
    float*          tmmid   = (float*)alloc((size_t)2 * MLPD * 4);
    float*          tmvec   = (float*)alloc((size_t)2 * DIMD * 4);

    // --- conv front-end + token packing ---
    conv1_kernel<<<(2 * 256 * 1024 + 255) / 256, 256, 0, stream>>>(x, conv_w1, conv_b1, conv1o);
    conv2_kernel<<<(2 * 512 * 1024 + 255) / 256, 256, 0, stream>>>(conv1o, conv_w2, conv_b2, conv2o);
    pack_kernel<<<(M * DIMD + 255) / 256, 256, 0, stream>>>(conv2o, cond, hbuf);

    // --- time embedding (M=2: scalar GEMMs) ---
    timefeat_kernel<<<(2 * DIMD + 255) / 256, 256, 0, stream>>>(tsteps, te);
    gemm_scalar_kernel<<<(2 * MLPD + 255) / 256, 256, 0, stream>>>(te, tpe_w1, tpe_b1, tpe_mid, 2, MLPD, DIMD, 1);
    gemm_scalar_kernel<<<(2 * DIMD + 255) / 256, 256, 0, stream>>>(tpe_mid, tpe_w2, tpe_b2, temb, 2, DIMD, MLPD, 0);

    // --- 12 transformer blocks ---
    for (int l = 0; l < 12; ++l) {
        const float* Wq  = wq     + (size_t)l * DIMD * DIMD;
        const float* Wkv = wkv    + (size_t)l * DIMD * 128;
        const float* Wo  = wo     + (size_t)l * DIMD * DIMD;
        const float* Bo  = bo     + (size_t)l * DIMD;
        const float* F1  = ffn_w1 + (size_t)l * DIMD * MLPD;
        const float* Fb1 = ffn_b1 + (size_t)l * MLPD;
        const float* F2  = ffn_w2 + (size_t)l * MLPD * DIMD;
        const float* Fb2 = ffn_b2 + (size_t)l * DIMD;
        const float* T1  = tm_w1  + (size_t)l * DIMD * MLPD;
        const float* Tb1 = tm_b1  + (size_t)l * MLPD;
        const float* T2  = tm_w2  + (size_t)l * MLPD * DIMD;
        const float* Tb2 = tm_b2  + (size_t)l * DIMD;

        // time modulation for this layer (M=2)
        gemm_scalar_kernel<<<(2 * MLPD + 255) / 256, 256, 0, stream>>>(temb, T1, Tb1, tmmid, 2, MLPD, DIMD, 1);
        gemm_scalar_kernel<<<(2 * DIMD + 255) / 256, 256, 0, stream>>>(tmmid, T2, Tb2, tmvec, 2, DIMD, MLPD, 0);

        // ln1 -> hx (f32 + bf16)
        layernorm_kernel<<<M, 128, 0, stream>>>(hbuf, ln1_g + l * DIMD, ln1_b + l * DIMD, hx, hx_bf, DIMD);

        // q / kv projections (WMMA, bf16 outputs)
        launch_wconv(Wq, wt_buf, DIMD, DIMD, stream);
        launch_gemm(hx_bf, wt_buf, nullptr, nullptr, nullptr, 0, nullptr, q_bf, M, DIMD, DIMD, 0, stream);
        launch_wconv(Wkv, wt_buf, DIMD, 128, stream);
        launch_gemm(hx_bf, wt_buf, nullptr, nullptr, nullptr, 0, nullptr, kv_bf, M, 128, DIMD, 0, stream);
        vtrans_kernel<<<(2 * 64 * NTOK + 255) / 256, 256, 0, stream>>>(kv_bf, vt_bf);

        // flash attention (WMMA)
        attention_kernel<<<dim3(NTOK / 16, 8, 2), 32, 0, stream>>>(q_bf, kv_bf, vt_bf, o_bf);

        // out proj + residual(hx) + time-mod broadcast -> h (WMMA)
        launch_wconv(Wo, wt_buf, DIMD, DIMD, stream);
        launch_gemm(o_bf, wt_buf, Bo, hx, tmvec, NTOK, hbuf, nullptr, M, DIMD, DIMD, 0, stream);

        // ln2 -> hx (f32 + bf16)
        layernorm_kernel<<<M, 128, 0, stream>>>(hbuf, ln2_g + l * DIMD, ln2_b + l * DIMD, hx, hx_bf, DIMD);

        // FFN: gelu(hx@F1+b1) -> bf16 mid; mid@F2+b2 + hx -> h (WMMA)
        launch_wconv(F1, wt_buf, DIMD, MLPD, stream);
        launch_gemm(hx_bf, wt_buf, Fb1, nullptr, nullptr, 0, nullptr, ffn_bf, M, MLPD, DIMD, 1, stream);
        launch_wconv(F2, wt_buf, MLPD, DIMD, stream);
        launch_gemm(ffn_bf, wt_buf, Fb2, hx, nullptr, 0, hbuf, nullptr, M, DIMD, MLPD, 0, stream);
    }

    // --- final norm + output head ---
    layernorm_kernel<<<M, 128, 0, stream>>>(hbuf, fn_g, fn_b, hx, hx_bf, DIMD);
    launch_wconv(out_w1, wt_buf, DIMD, 1024, stream);
    launch_gemm(hx_bf, wt_buf, out_b1, nullptr, nullptr, 0, nullptr, omid_bf, M, 1024, DIMD, 1, stream);
    final_out_kernel<<<(2 * 1024 + 255) / 256, 256, 0, stream>>>(omid_bf, out_w2, out_b2, (float*)d_out);
}
